// FinetuneBertFANAttention_61022895341685
// MI455X (gfx1250) — matrix operand order repaired and checked by reading, not compiled, not run on hardware
//
#include <hip/hip_runtime.h>

#define DDIM 768
#define BM 128
#define BN 128
#define BK 32
#define MROWS 32768
#define BATCH 64
#define SEQ 512
#define LN_EPS 1e-5f
#define COS_EPS 1e-8f

typedef __attribute__((ext_vector_type(16))) __bf16 v16bf;
typedef __attribute__((ext_vector_type(8)))  float  v8f;
typedef unsigned short ushort_t;

union BfVec { v16bf v; unsigned u[8]; };

// ---- helpers -------------------------------------------------------------

__device__ __forceinline__ void split1(float x, ushort_t& hi, ushort_t& lo) {
  __bf16 h = (__bf16)x;
  __bf16 l = (__bf16)(x - (float)h);
  union { __bf16 b; ushort_t s; } a, c;
  a.b = h; c.b = l;
  hi = a.s; lo = c.s;
}

__device__ __forceinline__ void split2(float x0, float x1, unsigned& hi, unsigned& lo) {
  ushort_t h0, l0, h1, l1;
  split1(x0, h0, l0);
  split1(x1, h1, l1);
  hi = (unsigned)h0 | ((unsigned)h1 << 16);
  lo = (unsigned)l0 | ((unsigned)l1 << 16);
}

// CDNA5 async global->LDS copy (16B per lane), tracked by ASYNCcnt.
__device__ __forceinline__ void async_copy16(unsigned lds_off, const void* gptr) {
  asm volatile("global_load_async_to_lds_b128 %0, %1, off"
               :: "v"(lds_off), "v"(gptr) : "memory");
}
__device__ __forceinline__ void wait_async0() {
  asm volatile("s_wait_asynccnt 0" ::: "memory");
}
__device__ __forceinline__ unsigned lds_off_of(const void* p) {
  return (unsigned)(size_t)p;   // generic->LDS: low 32 bits are the LDS offset
}

// ---- split-transpose of weight matrices: W[k][n] -> T{hi,lo}[n][k] -------

__global__ __launch_bounds__(256)
void split_transpose_kernel(const float* __restrict__ W,
                            ushort_t* __restrict__ Thi,
                            ushort_t* __restrict__ Tlo)
{
  __shared__ float t[32][33];
  const int tx = threadIdx.x & 31, ty = threadIdx.x >> 5;
  const int bk = blockIdx.x * 32, bn = blockIdx.y * 32;
  #pragma unroll
  for (int i = 0; i < 32; i += 8)
    t[ty + i][tx] = W[(size_t)(bk + ty + i) * DDIM + bn + tx];
  __syncthreads();
  #pragma unroll
  for (int i = 0; i < 32; i += 8) {
    ushort_t hi, lo;
    split1(t[tx][ty + i], hi, lo);
    size_t o = (size_t)(bn + ty + i) * DDIM + bk + tx;
    Thi[o] = hi; Tlo[o] = lo;
  }
}

// ---- GEMM: C = A[32768,768] x W[768,768] (+bias) -------------------------
// MODE 0: A from f32 (convert on the fly), epilogue relu -> bf16 hi/lo planes
// MODE 1: A from bf16 hi/lo planes (async copy), epilogue += residual -> f32
template<int MODE>
__global__ __launch_bounds__(256, 1)
void gemm_wmma_kernel(const float*   __restrict__ Af32,
                      const ushort_t* __restrict__ Ahi, const ushort_t* __restrict__ Alo,
                      const ushort_t* __restrict__ Bhi, const ushort_t* __restrict__ Blo,
                      const float*   __restrict__ bias,
                      const float*   __restrict__ R,
                      float*         __restrict__ Outf32,
                      ushort_t*      __restrict__ OutHi, ushort_t* __restrict__ OutLo)
{
  __shared__ unsigned sA[2][2][BM][BK / 2];   // [buf][hi/lo][row][kpair]
  __shared__ unsigned sB[2][2][BN][BK / 2];   // [buf][hi/lo][col][kpair]

  const int tid  = threadIdx.x;
  const int lane = tid & 31;
  const int wv   = tid >> 5;
  const int lr   = lane & 15;
  const int hf   = lane >> 4;
  const int wm   = (wv & 3) * 32;
  const int wn   = (wv >> 2) * 64;
  const size_t bm = (size_t)blockIdx.x * BM;
  const int    bn = blockIdx.y * BN;

  v8f acc[2][4] = {};
  float2 areg[8];

  // B tile: pure byte copy of WT rows (64B per row) via async-to-LDS
  auto stageB = [&](int buf, int kb) {
    #pragma unroll
    for (int i = 0; i < 2; ++i) {
      int c   = tid + i * 256;       // 0..511 : 128 rows x 4 16B-chunks
      int n   = c >> 2;
      int off = (c & 3) * 16;
      size_t gofs = ((size_t)(bn + n) * DDIM + kb) * 2 + off;
      async_copy16(lds_off_of(&sB[buf][0][n][0]) + off, (const char*)Bhi + gofs);
      async_copy16(lds_off_of(&sB[buf][1][n][0]) + off, (const char*)Blo + gofs);
    }
  };
  // A tile (MODE 1): byte copy of h1 hi/lo plane rows
  auto stageA_async = [&](int buf, int kb) {
    #pragma unroll
    for (int i = 0; i < 2; ++i) {
      int c   = tid + i * 256;
      int n   = c >> 2;
      int off = (c & 3) * 16;
      size_t gofs = ((bm + n) * DDIM + kb) * 2 + off;
      async_copy16(lds_off_of(&sA[buf][0][n][0]) + off, (const char*)Ahi + gofs);
      async_copy16(lds_off_of(&sA[buf][1][n][0]) + off, (const char*)Alo + gofs);
    }
  };
  // A tile (MODE 0): f32 loads into regs (issued early), convert+store later
  auto loadA = [&](int kb) {
    #pragma unroll
    for (int i = 0; i < 8; ++i) {
      int p = tid + i * 256; int row = p >> 4; int kp = p & 15;
      areg[i] = *(const float2*)(Af32 + (bm + row) * DDIM + kb + kp * 2);
    }
  };
  auto storeA = [&](int buf) {
    #pragma unroll
    for (int i = 0; i < 8; ++i) {
      int p = tid + i * 256; int row = p >> 4; int kp = p & 15;
      split2(areg[i].x, areg[i].y, sA[buf][0][row][kp], sA[buf][1][row][kp]);
    }
  };

  auto compute = [&](int buf) {
    BfVec aH[2], aL[2], bH[4], bL[4];
    #pragma unroll
    for (int mt = 0; mt < 2; ++mt) {
      int r = wm + mt * 16 + lr;
      #pragma unroll
      for (int v = 0; v < 8; ++v) {
        int k = ((v >> 2) << 4) + (hf << 3) + ((v & 3) << 1);
        aH[mt].u[v] = sA[buf][0][r][k >> 1];
        aL[mt].u[v] = sA[buf][1][r][k >> 1];
      }
    }
    #pragma unroll
    for (int nt = 0; nt < 4; ++nt) {
      int c = wn + nt * 16 + lr;
      #pragma unroll
      for (int v = 0; v < 8; ++v) {
        int k = ((v >> 2) << 4) + (hf << 3) + ((v & 3) << 1);
        bH[nt].u[v] = sB[buf][0][c][k >> 1];
        bL[nt].u[v] = sB[buf][1][c][k >> 1];
      }
    }
    #pragma unroll
    for (int mt = 0; mt < 2; ++mt) {
      #pragma unroll
      for (int nt = 0; nt < 4; ++nt) {
        acc[mt][nt] = __builtin_amdgcn_wmma_f32_16x16x32_bf16(
            false, aH[mt].v, false, bH[nt].v, (short)0, acc[mt][nt], false, false);
        acc[mt][nt] = __builtin_amdgcn_wmma_f32_16x16x32_bf16(
            false, aH[mt].v, false, bL[nt].v, (short)0, acc[mt][nt], false, false);
        acc[mt][nt] = __builtin_amdgcn_wmma_f32_16x16x32_bf16(
            false, aL[mt].v, false, bH[nt].v, (short)0, acc[mt][nt], false, false);
      }
    }
  };

  // prologue: fill buffer 0
  if (MODE == 0) { loadA(0); storeA(0); } else { stageA_async(0, 0); }
  stageB(0, 0);
  wait_async0();
  __syncthreads();

  // double-buffered main loop, 2 k-steps per iteration (buf indices constant)
  for (int kb = 0; kb < DDIM; kb += 2 * BK) {
    const bool m1 = (kb + BK) < DDIM;
    if (m1) {
      stageB(1, kb + BK);
      if (MODE == 1) stageA_async(1, kb + BK); else loadA(kb + BK);
    }
    compute(0);
    if (m1 && MODE == 0) storeA(1);
    wait_async0();
    __syncthreads();

    const bool m2 = (kb + 2 * BK) < DDIM;
    if (m2) {
      stageB(0, kb + 2 * BK);
      if (MODE == 1) stageA_async(0, kb + 2 * BK); else loadA(kb + 2 * BK);
    }
    compute(1);
    if (m2 && MODE == 0) storeA(0);
    wait_async0();
    __syncthreads();
  }

  // epilogue: C layout -> row = base + r + 8*half, col = base + lane%16
  #pragma unroll
  for (int mt = 0; mt < 2; ++mt) {
    #pragma unroll
    for (int nt = 0; nt < 4; ++nt) {
      int col = bn + wn + nt * 16 + lr;
      float bc = bias[col];
      #pragma unroll
      for (int r = 0; r < 8; ++r) {
        size_t row = bm + wm + mt * 16 + hf * 8 + r;
        float vv = acc[mt][nt][r] + bc;
        if (MODE == 0) {
          vv = fmaxf(vv, 0.0f);
          ushort_t hi, lo;
          split1(vv, hi, lo);
          OutHi[row * DDIM + col] = hi;
          OutLo[row * DDIM + col] = lo;
        } else {
          vv += R[row * DDIM + col];
          Outf32[row * DDIM + col] = vv;
        }
      }
    }
  }
}

// ---- LayerNorm + tiny projections + per-row inverse norm -----------------

__device__ __forceinline__ float block_reduce_sum(float v, float* red, int tid) {
  red[tid] = v;
  __syncthreads();
  for (int off = 128; off > 0; off >>= 1) {
    if (tid < off) red[tid] += red[tid + off];
    __syncthreads();
  }
  float r = red[0];
  __syncthreads();
  return r;
}

__global__ __launch_bounds__(256)
void ln_proj_kernel(float* __restrict__ X,
                    const float* __restrict__ gam, const float* __restrict__ bet,
                    const float* __restrict__ Wy, const float* __restrict__ by,
                    const float* __restrict__ Wz, const float* __restrict__ bz,
                    float* __restrict__ out1, float* __restrict__ out2,
                    float* __restrict__ invnorm)
{
  __shared__ float red[256];
  const int tid = threadIdx.x;
  const size_t row = blockIdx.x;
  float* xr = X + row * DDIM;

  float x[3]; float s = 0.f, ss = 0.f;
  #pragma unroll
  for (int i = 0; i < 3; ++i) {
    x[i] = xr[tid + i * 256];
    s += x[i]; ss += x[i] * x[i];
  }
  s  = block_reduce_sum(s,  red, tid);
  ss = block_reduce_sum(ss, red, tid);
  float mean = s * (1.0f / DDIM);
  float var  = ss * (1.0f / DDIM) - mean * mean;
  float rstd = rsqrtf(var + LN_EPS);

  float ffsq = 0.f;
  float py[3] = {0, 0, 0}, pz[5] = {0, 0, 0, 0, 0};
  #pragma unroll
  for (int i = 0; i < 3; ++i) {
    int d = tid + i * 256;
    float ff = gam[d] * (x[i] - mean) * rstd + bet[d];
    xr[d] = ff;
    ffsq += ff * ff;
    #pragma unroll
    for (int j = 0; j < 3; ++j) py[j] += ff * Wy[d * 3 + j];
    #pragma unroll
    for (int j = 0; j < 5; ++j) pz[j] += ff * Wz[d * 5 + j];
  }
  ffsq = block_reduce_sum(ffsq, red, tid);
  #pragma unroll
  for (int j = 0; j < 3; ++j) py[j] = block_reduce_sum(py[j], red, tid);
  #pragma unroll
  for (int j = 0; j < 5; ++j) pz[j] = block_reduce_sum(pz[j], red, tid);

  if (tid == 0) {
    invnorm[row] = 1.0f / fmaxf(sqrtf(ffsq), COS_EPS);
    #pragma unroll
    for (int j = 0; j < 3; ++j) out1[row * 3 + j] = py[j] + by[j];
    #pragma unroll
    for (int j = 0; j < 5; ++j) out2[row * 5 + j] = pz[j] + bz[j];
  }
}

// ---- per-batch sum of normalized rows ------------------------------------

__global__ __launch_bounds__(256)
void batch_sum_kernel(const float* __restrict__ FF, const float* __restrict__ invnorm,
                      float* __restrict__ S)
{
  const int b = blockIdx.x, tid = threadIdx.x;
  float a0 = 0.f, a1 = 0.f, a2 = 0.f;
  for (int l = 0; l < SEQ; ++l) {
    float inv = invnorm[b * SEQ + l];
    const float* r = FF + (size_t)(b * SEQ + l) * DDIM;
    a0 += r[tid      ] * inv;
    a1 += r[tid + 256] * inv;
    a2 += r[tid + 512] * inv;
  }
  S[b * DDIM + tid      ] = a0;
  S[b * DDIM + tid + 256] = a1;
  S[b * DDIM + tid + 512] = a2;
}

// ---- anti-sigmoid attention softmax --------------------------------------

__global__ __launch_bounds__(256)
void attention_kernel(const float* __restrict__ FF, const float* __restrict__ invnorm,
                      const float* __restrict__ S, float* __restrict__ attn)
{
  __shared__ float sS[DDIM];
  __shared__ float logit[SEQ];
  __shared__ float red[256];
  const int b = blockIdx.x, tid = threadIdx.x;
  #pragma unroll
  for (int i = 0; i < 3; ++i) sS[tid + i * 256] = S[b * DDIM + tid + i * 256];
  __syncthreads();

  for (int l = tid; l < SEQ; l += 256) {
    const float4* r = (const float4*)(FF + (size_t)(b * SEQ + l) * DDIM);
    float dot = 0.f;
    #pragma unroll 4
    for (int d = 0; d < DDIM / 4; ++d) {
      float4 v = r[d];
      dot += v.x * sS[d * 4] + v.y * sS[d * 4 + 1] + v.z * sS[d * 4 + 2] + v.w * sS[d * 4 + 3];
    }
    float aw = dot * invnorm[b * SEQ + l] * (1.0f / SEQ);
    logit[l] = 1.0f / (1.0f + expf(aw));
  }
  __syncthreads();

  float m = fmaxf(logit[tid], logit[tid + 256]);
  red[tid] = m; __syncthreads();
  for (int off = 128; off > 0; off >>= 1) {
    if (tid < off) red[tid] = fmaxf(red[tid], red[tid + off]);
    __syncthreads();
  }
  float mx = red[0]; __syncthreads();

  float e0 = expf(logit[tid] - mx);
  float e1 = expf(logit[tid + 256] - mx);
  red[tid] = e0 + e1; __syncthreads();
  for (int off = 128; off > 0; off >>= 1) {
    if (tid < off) red[tid] += red[tid + off];
    __syncthreads();
  }
  float inv = 1.0f / red[0];
  attn[b * SEQ + tid      ] = e0 * inv;
  attn[b * SEQ + tid + 256] = e1 * inv;
}

// ---- launch --------------------------------------------------------------

extern "C" void kernel_launch(void* const* d_in, const int* in_sizes, int n_in,
                              void* d_out, int out_size, void* d_ws, size_t ws_size,
                              hipStream_t stream)
{
  (void)in_sizes; (void)n_in; (void)out_size; (void)ws_size;
  const float* E   = (const float*)d_in[0];
  const float* W1  = (const float*)d_in[1];
  const float* b1  = (const float*)d_in[2];
  const float* W2  = (const float*)d_in[3];
  const float* b2  = (const float*)d_in[4];
  const float* gam = (const float*)d_in[5];
  const float* bet = (const float*)d_in[6];
  const float* Wy  = (const float*)d_in[7];
  const float* by  = (const float*)d_in[8];
  const float* Wz  = (const float*)d_in[9];
  const float* bz  = (const float*)d_in[10];

  float* out1 = (float*)d_out;                     // [64,512,3]
  float* out2 = out1 + (size_t)BATCH * SEQ * 3;    // [64,512,5]
  float* attn = out2 + (size_t)BATCH * SEQ * 5;    // [64,512]

  // workspace layout
  char* ws = (char*)d_ws;
  const size_t NE = (size_t)MROWS * DDIM;          // 25165824
  const size_t NW = (size_t)DDIM * DDIM;           // 589824
  float*    xff   = (float*)ws;            ws += NE * 4;   // x -> ff in place
  ushort_t* H1hi  = (ushort_t*)ws;         ws += NE * 2;
  ushort_t* H1lo  = (ushort_t*)ws;         ws += NE * 2;
  ushort_t* W1Thi = (ushort_t*)ws;         ws += NW * 2;
  ushort_t* W1Tlo = (ushort_t*)ws;         ws += NW * 2;
  ushort_t* W2Thi = (ushort_t*)ws;         ws += NW * 2;
  ushort_t* W2Tlo = (ushort_t*)ws;         ws += NW * 2;
  float*    invn  = (float*)ws;            ws += (size_t)MROWS * 4;
  float*    S     = (float*)ws;

  dim3 gt(DDIM / 32, DDIM / 32);
  split_transpose_kernel<<<gt, 256, 0, stream>>>(W1, W1Thi, W1Tlo);
  split_transpose_kernel<<<gt, 256, 0, stream>>>(W2, W2Thi, W2Tlo);

  dim3 gg(MROWS / BM, DDIM / BN);
  gemm_wmma_kernel<0><<<gg, 256, 0, stream>>>(E, nullptr, nullptr, W1Thi, W1Tlo,
                                              b1, nullptr, nullptr, H1hi, H1lo);
  gemm_wmma_kernel<1><<<gg, 256, 0, stream>>>(nullptr, H1hi, H1lo, W2Thi, W2Tlo,
                                              b2, E, xff, nullptr, nullptr);

  ln_proj_kernel<<<MROWS, 256, 0, stream>>>(xff, gam, bet, Wy, by, Wz, bz,
                                            out1, out2, invn);
  batch_sum_kernel<<<BATCH, 256, 0, stream>>>(xff, invn, S);
  attention_kernel<<<BATCH, 256, 0, stream>>>(xff, invn, S, attn);
}
